// Bengio03ResNetBiLm_21345987461678
// MI455X (gfx1250) — matrix operand order repaired and checked by reading, not compiled, not run on hardware
//
#include <hip/hip_runtime.h>
#include <hip/hip_bf16.h>
#include <stdint.h>

// ---------------------------------------------------------------------------
// Problem constants (match reference)
// ---------------------------------------------------------------------------
constexpr int Bc = 32;
constexpr int Sc = 2048;
constexpr int Hc = 256;
constexpr int Wc = 3;
constexpr int Lc = 3;
constexpr int SP = Sc + 2 * Wc;        // 2054 padded time steps
constexpr int Mrows = Bc * Sc;         // 65536 output rows per GEMM
constexpr int CTXK = Hc * (Wc + 1);    // 1024 context K

typedef __bf16 bf16_t;
typedef bf16_t bf16x8 __attribute__((ext_vector_type(8)));
typedef bf16_t v16bf  __attribute__((ext_vector_type(16)));
typedef float  v8f    __attribute__((ext_vector_type(8)));

// ---------------------------------------------------------------------------
// Helpers
// ---------------------------------------------------------------------------
__device__ __host__ inline bf16_t f2bf(float f) {
    union { float f; unsigned u; } in; in.f = f;
    unsigned u = in.u;
    u += 0x7FFFu + ((u >> 16) & 1u);          // round-to-nearest-even
    union { unsigned short s; bf16_t b; } out;
    out.s = (unsigned short)(u >> 16);
    return out.b;
}

// A-matrix fragment (16x32 bf16). Per ISA 7.12.2: lanes 0-15 -> M=0..15 hold
// K {0..7, 16..23}; lanes 16-31 hold K {8..15, 24..31}.
__device__ inline v16bf load_frag_a(const bf16_t* __restrict__ base, int lda, int lane) {
    int r = lane & 15;
    int h = lane >> 4;
    const bf16_t* p = base + (size_t)r * lda + h * 8;
    union { v16bf v; bf16x8 q[2]; } f;
    f.q[0] = *reinterpret_cast<const bf16x8*>(p);        // K = k0 + h*8 .. +7
    f.q[1] = *reinterpret_cast<const bf16x8*>(p + 16);   // K = k0 + 16 + h*8 ..
    return f.v;
}

// B-matrix fragment (32x16 bf16), loaded from B^T stored [N,K] row-major.
// Same K order as A: lanes 0-15 hold K {0..7, 16..23} of col n,
// lanes 16-31 hold K {8..15, 24..31} of col n.
__device__ inline v16bf load_frag_b(const bf16_t* __restrict__ rowbase, int ldb, int lane) {
    int n = lane & 15;
    int h = lane >> 4;
    const bf16_t* p = rowbase + (size_t)n * ldb + h * 8;
    union { v16bf v; bf16x8 q[2]; } f;
    f.q[0] = *reinterpret_cast<const bf16x8*>(p);
    f.q[1] = *reinterpret_cast<const bf16x8*>(p + 16);
    return f.v;
}

__device__ inline v8f wmma_bf16(v16bf a, v16bf b, v8f c) {
    // (neg_a, A, neg_b, B, c_mod, C, reuse_a, reuse_b)
    return __builtin_amdgcn_wmma_f32_16x16x32_bf16(false, a, false, b, (short)0, c,
                                                   false, false);
}

// ---------------------------------------------------------------------------
// Weight convert + transpose: src [layers, K, N] f32  ->  dst [layers, N, K] bf16
// ---------------------------------------------------------------------------
__global__ void tconv_kernel(const float* __restrict__ src, bf16_t* __restrict__ dst,
                             int K, int N, int layers) {
    size_t idx = (size_t)blockIdx.x * blockDim.x + threadIdx.x;
    size_t per = (size_t)K * N;
    size_t total = per * layers;
    if (idx >= total) return;
    size_t l = idx / per;
    size_t r = idx - l * per;
    int n = (int)(r / K);
    int k = (int)(r - (size_t)n * K);
    dst[idx] = f2bf(src[l * per + (size_t)k * N + n]);
}

// ---------------------------------------------------------------------------
// Build padded bf16 activations: out [B, SP, H] = [front pad | x | back pad]
// Each thread handles 8 consecutive channels (16-byte bf16 store).
// ---------------------------------------------------------------------------
__global__ void build_padded_kernel(const float* __restrict__ x,
                                    const float* __restrict__ fp,   // [W,H]
                                    const float* __restrict__ bp,   // [W,H]
                                    bf16_t* __restrict__ out) {
    int idx = blockIdx.x * blockDim.x + threadIdx.x;
    constexpr int C8 = Hc / 8;
    int total = Bc * SP * C8;
    if (idx >= total) return;
    int c   = (idx % C8) * 8;
    int pos = (idx / C8) % SP;
    int b   = idx / (C8 * SP);
    const float* src;
    if (pos < Wc)            src = fp + (size_t)pos * Hc + c;
    else if (pos >= Sc + Wc) src = bp + (size_t)(pos - Sc - Wc) * Hc + c;
    else                     src = x + ((size_t)b * Sc + (pos - Wc)) * Hc + c;
    float4 lo = reinterpret_cast<const float4*>(src)[0];
    float4 hi = reinterpret_cast<const float4*>(src)[1];
    bf16x8 o;
    o[0] = f2bf(lo.x); o[1] = f2bf(lo.y); o[2] = f2bf(lo.z); o[3] = f2bf(lo.w);
    o[4] = f2bf(hi.x); o[5] = f2bf(hi.y); o[6] = f2bf(hi.z); o[7] = f2bf(hi.w);
    *reinterpret_cast<bf16x8*>(out + ((size_t)b * SP + pos) * Hc + c) = o;
}

// ---------------------------------------------------------------------------
// Row-wise LayerNorm (H=256): one wave per row, 8 f32 per lane, wave32
// shuffle reduction. Output bf16 for the FFN GEMM.
// ---------------------------------------------------------------------------
__global__ __launch_bounds__(256) void layernorm_kernel(const float* __restrict__ Hm,
                                                        const float* __restrict__ g,
                                                        const float* __restrict__ be,
                                                        bf16_t* __restrict__ Y) {
    int wave = threadIdx.x >> 5;
    int lane = threadIdx.x & 31;
    int row  = blockIdx.x * 8 + wave;
    const float* p = Hm + (size_t)row * Hc + lane * 8;
    float4 a = reinterpret_cast<const float4*>(p)[0];
    float4 b = reinterpret_cast<const float4*>(p)[1];
    float v[8] = {a.x, a.y, a.z, a.w, b.x, b.y, b.z, b.w};
    float s = 0.f, sq = 0.f;
#pragma unroll
    for (int r = 0; r < 8; ++r) { s += v[r]; sq += v[r] * v[r]; }
#pragma unroll
    for (int off = 16; off > 0; off >>= 1) {
        s  += __shfl_xor(s,  off, 32);
        sq += __shfl_xor(sq, off, 32);
    }
    float mean = s * (1.f / Hc);
    float var  = sq * (1.f / Hc) - mean * mean;
    float inv  = rsqrtf(var + 1e-6f);
    bf16x8 o;
#pragma unroll
    for (int r = 0; r < 8; ++r) {
        int c = lane * 8 + r;
        o[r] = f2bf(g[c] * (v[r] - mean) * inv + be[c]);
    }
    *reinterpret_cast<bf16x8*>(Y + (size_t)row * Hc + lane * 8) = o;
}

// ---------------------------------------------------------------------------
// bf16 WMMA GEMM, f32 accumulate.  D[M,256] = A[M,K] * Bt[256,K]^T + bias
//   workgroup = 8 waves (4M x 2N), tile 128(M) x 128(N)
//   wave      = 32(M) x 64(N) = 2x4 WMMA tiles -> 8 WMMAs per 12 b128 loads
// MODE 0: ctx GEMM  (overlapping-row A addressing) -> Fout = relu(.) f32
// MODE 1: FFN1      -> Ubf = relu(.) bf16
// MODE 2: FFN2      -> Fout += (.)  (residual, in place) and Dout[m*512+n]
// ---------------------------------------------------------------------------
template <int MODE>
__global__ __launch_bounds__(256) void gemm_kernel(const bf16_t* __restrict__ A,
                                                   const bf16_t* __restrict__ Bt,
                                                   const float* __restrict__ bias,
                                                   float* Fout,
                                                   bf16_t* __restrict__ Ubf,
                                                   float* __restrict__ Dout,
                                                   int K, int lda, int ctx_off) {
    int wave = threadIdx.x >> 5;
    int lane = threadIdx.x & 31;
    int wm = wave >> 1;            // 0..3  (M direction)
    int wn = wave & 1;             // 0..1  (N direction)
    int m0 = blockIdx.x * 128 + wm * 32;
    int n0 = blockIdx.y * 128 + wn * 64;

    const bf16_t* Aw;
    if (MODE == 0) {
        // context GEMM: A row m=(b,t) is a 1024-long slice of padded[b] starting
        // at time (t + ctx_off), row stride H. 128-row tiles never cross b.
        int mblk = blockIdx.x * 128;
        int bb = mblk >> 11;           // / Sc
        int t0 = mblk & (Sc - 1);
        Aw = A + ((size_t)bb * SP + t0 + ctx_off) * Hc + (size_t)(wm * 32) * lda;
    } else {
        Aw = A + (size_t)m0 * lda;
    }
    const bf16_t* Bw0 = Bt + (size_t)n0 * K;

    v8f acc[2][4] = {};
    for (int k = 0; k < K; k += 32) {
        v16bf a0 = load_frag_a(Aw + k, lda, lane);
        v16bf a1 = load_frag_a(Aw + (size_t)16 * lda + k, lda, lane);
        v16bf bf0 = load_frag_b(Bw0 + k, K, lane);
        v16bf bf1 = load_frag_b(Bw0 + (size_t)16 * K + k, K, lane);
        v16bf bf2 = load_frag_b(Bw0 + (size_t)32 * K + k, K, lane);
        v16bf bf3 = load_frag_b(Bw0 + (size_t)48 * K + k, K, lane);
        acc[0][0] = wmma_bf16(a0, bf0, acc[0][0]);
        acc[1][0] = wmma_bf16(a1, bf0, acc[1][0]);
        acc[0][1] = wmma_bf16(a0, bf1, acc[0][1]);
        acc[1][1] = wmma_bf16(a1, bf1, acc[1][1]);
        acc[0][2] = wmma_bf16(a0, bf2, acc[0][2]);
        acc[1][2] = wmma_bf16(a1, bf2, acc[1][2]);
        acc[0][3] = wmma_bf16(a0, bf3, acc[0][3]);
        acc[1][3] = wmma_bf16(a1, bf3, acc[1][3]);
    }

    // D layout (ISA 7.12.2): VGPR r, lanes 0-15: M=r, N=lane; lanes 16-31: M=8+r.
    int nc = lane & 15;
    int hf = lane >> 4;
#pragma unroll
    for (int sm = 0; sm < 2; ++sm) {
        int mbase = m0 + sm * 16 + hf * 8;
#pragma unroll
        for (int sn = 0; sn < 4; ++sn) {
            int n = n0 + sn * 16 + nc;
            float bv = bias[n];
#pragma unroll
            for (int r = 0; r < 8; ++r) {
                int m = mbase + r;
                float vv = acc[sm][sn][r] + bv;
                size_t o = (size_t)m * Hc + n;
                if (MODE == 0) {
                    Fout[o] = fmaxf(vv, 0.f);
                } else if (MODE == 1) {
                    Ubf[o] = f2bf(fmaxf(vv, 0.f));
                } else {
                    float res = Fout[o] + vv;         // residual h + ffn (in place)
                    Fout[o] = res;
                    Dout[(size_t)m * (2 * Hc) + n] = res;
                }
            }
        }
    }
}

// ---------------------------------------------------------------------------
// Host orchestration
// ---------------------------------------------------------------------------
extern "C" void kernel_launch(void* const* d_in, const int* in_sizes, int n_in,
                              void* d_out, int out_size, void* d_ws, size_t ws_size,
                              hipStream_t stream) {
    (void)in_sizes; (void)n_in; (void)out_size; (void)ws_size;

    const float* x_in    = (const float*)d_in[0];
    const float* fwd_pad = (const float*)d_in[1];
    const float* bwd_pad = (const float*)d_in[2];
    const float* ctxW[2] = {(const float*)d_in[3], (const float*)d_in[5]};
    const float* ctxB[2] = {(const float*)d_in[4], (const float*)d_in[6]};
    const float* ln_g[2] = {(const float*)d_in[7], (const float*)d_in[9]};
    const float* ln_b[2] = {(const float*)d_in[8], (const float*)d_in[10]};
    const float* w1[2]   = {(const float*)d_in[11], (const float*)d_in[15]};
    const float* b1[2]   = {(const float*)d_in[12], (const float*)d_in[16]};
    const float* w2[2]   = {(const float*)d_in[13], (const float*)d_in[17]};
    const float* b2[2]   = {(const float*)d_in[14], (const float*)d_in[18]};
    float* out = (float*)d_out;

    // ---- workspace carve-up -------------------------------------------------
    char* ws = (char*)d_ws;
    size_t off = 0;
    auto carve = [&](size_t bytes) -> void* {
        void* p = ws + off;
        off = (off + bytes + 255) & ~(size_t)255;
        return p;
    };
    size_t szPadded = (size_t)Bc * SP * Hc * sizeof(bf16_t);
    size_t szH      = (size_t)Mrows * Hc * sizeof(float);
    size_t szY      = (size_t)Mrows * Hc * sizeof(bf16_t);
    size_t szWctx   = (size_t)Lc * Hc * CTXK * sizeof(bf16_t);
    size_t szWffn   = (size_t)Lc * Hc * Hc * sizeof(bf16_t);

    bf16_t* padded[2] = {(bf16_t*)carve(szPadded), (bf16_t*)carve(szPadded)};
    float*  hbuf[2]   = {(float*)carve(szH), (float*)carve(szH)};
    bf16_t* ybuf[2]   = {(bf16_t*)carve(szY), (bf16_t*)carve(szY)};
    bf16_t* Wt[2]     = {(bf16_t*)carve(szWctx), (bf16_t*)carve(szWctx)};
    bf16_t* w1t[2]    = {(bf16_t*)carve(szWffn), (bf16_t*)carve(szWffn)};
    bf16_t* w2t[2]    = {(bf16_t*)carve(szWffn), (bf16_t*)carve(szWffn)};
    bf16_t* ubuf[2]   = {padded[0], padded[1]};   // reuse: padded dead after ctx GEMM

    // ---- weight conversion (transpose to [N,K] bf16) ------------------------
    {
        int tC = (int)(((size_t)Lc * CTXK * Hc + 255) / 256);
        int tF = (int)(((size_t)Lc * Hc * Hc + 255) / 256);
        for (int br = 0; br < 2; ++br) {
            tconv_kernel<<<tC, 256, 0, stream>>>(ctxW[br], Wt[br], CTXK, Hc, Lc);
            tconv_kernel<<<tF, 256, 0, stream>>>(w1[br],  w1t[br], Hc, Hc, Lc);
            tconv_kernel<<<tF, 256, 0, stream>>>(w2[br],  w2t[br], Hc, Hc, Lc);
        }
    }

    dim3 ggrid(Mrows / 128, Hc / 128);           // (512, 2)
    int padBlocks = (Bc * SP * (Hc / 8) + 255) / 256;

    for (int i = 0; i < Lc; ++i) {
        for (int br = 0; br < 2; ++br) {
            const float* xsrc = (i == 0) ? x_in : hbuf[br];
            // [front fwd_pad[i] | x | back bwd_pad[i]]  (both branches pad identically)
            build_padded_kernel<<<padBlocks, 256, 0, stream>>>(
                xsrc, fwd_pad + (size_t)i * Wc * Hc, bwd_pad + (size_t)i * Wc * Hc,
                padded[br]);

            // h = relu(ctx @ Wp + b) ; fwd offset 0, bwd offset W
            gemm_kernel<0><<<ggrid, 256, 0, stream>>>(
                padded[br], Wt[br] + (size_t)i * Hc * CTXK, ctxB[br] + (size_t)i * Hc,
                hbuf[br], nullptr, nullptr, CTXK, Hc, (br == 0) ? 0 : Wc);

            // y = LN(h)
            layernorm_kernel<<<Mrows / 8, 256, 0, stream>>>(
                hbuf[br], ln_g[br] + (size_t)i * Hc, ln_b[br] + (size_t)i * Hc, ybuf[br]);

            // u = relu(y @ w1 + b1)
            gemm_kernel<1><<<ggrid, 256, 0, stream>>>(
                ybuf[br], w1t[br] + (size_t)i * Hc * Hc, b1[br] + (size_t)i * Hc,
                nullptr, ubuf[br], nullptr, Hc, Hc, 0);

            // x_next = h + (u @ w2 + b2); also write out[i, :, :, br*H : br*H+H]
            gemm_kernel<2><<<ggrid, 256, 0, stream>>>(
                ubuf[br], w2t[br] + (size_t)i * Hc * Hc, b2[br] + (size_t)i * Hc,
                hbuf[br], nullptr,
                out + (size_t)i * Mrows * (2 * Hc) + (size_t)br * Hc,
                Hc, Hc, 0);
        }
    }
}
